// DigitCaps_37881611551097
// MI455X (gfx1250) — matrix operand order repaired and checked by reading, not compiled, not run on hardware
//
#include <hip/hip_runtime.h>
#include <hip/hip_bf16.h>

typedef __attribute__((ext_vector_type(16))) __bf16 v16bf;
typedef __attribute__((ext_vector_type(8)))  __bf16 v8bf;
typedef __attribute__((ext_vector_type(2)))  __bf16 v2bf;
typedef __attribute__((ext_vector_type(8)))  float  v8f;

#define Bsz   128
#define ICn   64
#define Rn    1152
#define Cn    16
#define On    64
#define Ncol  1024        // C*O
#define RCHUNK 18
#define RCHUNKS 64        // 64*18 = 1152
#define AROW  72          // padded bf16 row stride (144B, 16B aligned, conflict-free)

// ---------------------------------------------------------------- transpose
// x (B, IC, R) -> xT (B, R, IC) so A tiles are contiguous in i.
__global__ void transpose_x(const float* __restrict__ x, float* __restrict__ xT) {
    __shared__ float tile[32][33];
    const int b  = blockIdx.z;
    const int i0 = blockIdx.y * 32;
    const int r0 = blockIdx.x * 32;
    const int tx = threadIdx.x, ty = threadIdx.y;
    for (int k = 0; k < 32; k += 8)
        tile[ty + k][tx] = x[((size_t)b * ICn + (i0 + ty + k)) * Rn + (r0 + tx)];
    __syncthreads();
    for (int k = 0; k < 32; k += 8)
        xT[((size_t)b * Rn + (r0 + ty + k)) * ICn + (i0 + tx)] = tile[tx][ty + k];
}

// ---------------------------------------------------------------- zero s
__global__ void zero_s(float* __restrict__ s) {
    s[blockIdx.x * 256 + threadIdx.x] = 0.0f;
}

// ---------------------------------------------------------------- squash
// Elementwise (reference squashes over a size-1 axis). Also re-zeroes s.
__global__ void squash_k(float* __restrict__ s, float* __restrict__ dst) {
    const int t = blockIdx.x * 256 + threadIdx.x;
    const float sv = s[t];
    const float sq = sv * sv;
    dst[t] = sq / (1.0f + sq) * sv * rsqrtf(sq + 1e-8f);
    s[t] = 0.0f;
}

// ---------------------------------------------------------------- routing pass
// One fused pass over r: recompute u_hat tiles with WMMA bf16, do agreement +
// softmax on the fly, accumulate s in registers, atomically flush at end.
// grid = (8 b-tiles, 64 r-chunks), block = 512 (16 waves; wave w <-> capsule c=w)
__launch_bounds__(512)
__global__ void routing_pass(const float* __restrict__ xT, const float* __restrict__ W,
                             float* __restrict__ b_ij, float* __restrict__ s,
                             const float* __restrict__ v_prev, int pass)
{
    __shared__ __attribute__((aligned(16))) __bf16 aTile[16 * AROW];
    __shared__ float a_lds[16 * 16];   // a_lds[c][m]
    __shared__ float c_lds[16 * 16];   // c_lds[c][m]

    const int tid  = threadIdx.x;
    const int wave = tid >> 5;          // 0..15 == capsule c
    const int lane = tid & 31;
    const int nl   = lane & 15;         // N / column-in-tile index
    const int half = lane >> 4;         // lane half (M-group / K-group select)
    const int bt   = blockIdx.x;        // b-tile 0..7
    const int r0   = blockIdx.y * RCHUNK;

    // v_prev for this wave's capsule: o = t*16 + nl
    float vv[4] = {0.f, 0.f, 0.f, 0.f};
    if (pass > 0) {
        for (int t = 0; t < 4; ++t)
            vv[t] = v_prev[wave * On + t * 16 + nl];
    }

    v8f s_acc[4];
    for (int t = 0; t < 4; ++t)
        for (int j = 0; j < 8; ++j) s_acc[t][j] = 0.0f;

    // staging coords: thread -> (row m, packed i-pair)
    const int sm = tid >> 5;            // 0..15
    const int si = (tid & 31) * 2;      // 0..62

    for (int rr = 0; rr < RCHUNK; ++rr) {
        const int r = r0 + rr;

        // ---- stage A (x tile, 16 b x 64 i) to LDS as packed bf16 pairs ----
        __syncthreads();
        {
            const float2 xv = *(const float2*)
                &xT[((size_t)(bt * 16 + sm) * Rn + r) * ICn + si];
            v2bf p; p[0] = (__bf16)xv.x; p[1] = (__bf16)xv.y;
            *(v2bf*)&aTile[sm * AROW + si] = p;      // one ds_store_b32
        }
        // ---- prefetch next r's W slice for this wave (128 x 128B lines) ----
        if (rr + 1 < RCHUNK) {
            const char* pb = (const char*)(W + (size_t)(r + 1) * 65536)
                             + wave * 16384;
            for (int q = 0; q < 4; ++q)
                __builtin_prefetch(pb + (q * 32 + lane) * 128, 0, 3);
        }
        __syncthreads();

        // ---- A fragments (contiguous k = 16*half + s, same order as B), 2 k-steps ----
        v16bf afrag[2];
        for (int ks = 0; ks < 2; ++ks) {
            const __bf16* base = &aTile[nl * AROW + 32 * ks + 16 * half];
            const v8bf lo = *(const v8bf*)base;
            const v8bf hi = *(const v8bf*)(base + 8);
            for (int e = 0; e < 8; ++e) { afrag[ks][e] = lo[e]; afrag[ks][8 + e] = hi[e]; }
        }

        // ---- u_hat tiles via WMMA; fold agreement partials ----
        v8f uh[4];
        float acc_a[8];
        for (int j = 0; j < 8; ++j) acc_a[j] = 0.0f;

        for (int t = 0; t < 4; ++t) {
            v8f acc = {};
            const int n = wave * 64 + t * 16 + nl;           // column in (c,o)
            for (int ks = 0; ks < 2; ++ks) {
                const float* wrow = W + (size_t)r * 65536 + (size_t)n * 64
                                      + 32 * ks + 16 * half; // 16 consecutive i
                const float4 f0 = *(const float4*)(wrow);
                const float4 f1 = *(const float4*)(wrow + 4);
                const float4 f2 = *(const float4*)(wrow + 8);
                const float4 f3 = *(const float4*)(wrow + 12);
                v16bf bfrag;
                bfrag[0]=(__bf16)f0.x;  bfrag[1]=(__bf16)f0.y;
                bfrag[2]=(__bf16)f0.z;  bfrag[3]=(__bf16)f0.w;
                bfrag[4]=(__bf16)f1.x;  bfrag[5]=(__bf16)f1.y;
                bfrag[6]=(__bf16)f1.z;  bfrag[7]=(__bf16)f1.w;
                bfrag[8]=(__bf16)f2.x;  bfrag[9]=(__bf16)f2.y;
                bfrag[10]=(__bf16)f2.z; bfrag[11]=(__bf16)f2.w;
                bfrag[12]=(__bf16)f3.x; bfrag[13]=(__bf16)f3.y;
                bfrag[14]=(__bf16)f3.z; bfrag[15]=(__bf16)f3.w;
                acc = __builtin_amdgcn_wmma_f32_16x16x32_bf16(
                          false, afrag[ks], false, bfrag, (short)0, acc, false, false);
            }
            uh[t] = acc;
            if (pass > 0) {
                const float vo = vv[t];
                for (int j = 0; j < 8; ++j) acc_a[j] += acc[j] * vo;
            }
        }

        if (pass > 0) {
            // ---- agreement: reduce over o (16 lanes of each half) ----
            for (int mask = 1; mask < 16; mask <<= 1)
                for (int j = 0; j < 8; ++j)
                    acc_a[j] += __shfl_xor(acc_a[j], mask, 32);
            if (nl == 0)
                for (int j = 0; j < 8; ++j)
                    a_lds[wave * 16 + half * 8 + j] = acc_a[j];   // a[c][m]
            __syncthreads();

            // ---- b update + softmax over C (threads 0..255: one (m,c)) ----
            if (tid < 256) {
                const int m = tid >> 4, ci = tid & 15;
                float bv = a_lds[ci * 16 + m];
                const size_t bix = ((size_t)(bt * 16 + m) * Rn + r) * Cn + ci;
                if (pass == 2) bv += b_ij[bix];
                if (pass == 1) b_ij[bix] = bv;       // b1 = a0 (b0 == 0)
                float mx = bv;
                for (int mask = 1; mask < 16; mask <<= 1)
                    mx = fmaxf(mx, __shfl_xor(mx, mask, 32));
                const float e = __expf(bv - mx);
                float sum = e;
                for (int mask = 1; mask < 16; mask <<= 1)
                    sum += __shfl_xor(sum, mask, 32);
                c_lds[ci * 16 + m] = e / sum;
            }
            __syncthreads();

            // ---- s += c * u_hat (c row for this wave's capsule) ----
            const float* crow = &c_lds[wave * 16 + half * 8];
            const float4 c0 = *(const float4*)crow;
            const float4 c1 = *(const float4*)(crow + 4);
            const float cj[8] = {c0.x, c0.y, c0.z, c0.w, c1.x, c1.y, c1.z, c1.w};
            for (int t = 0; t < 4; ++t)
                for (int j = 0; j < 8; ++j)
                    s_acc[t][j] += cj[j] * uh[t][j];
        } else {
            // pass 0: c is uniform 1/16 -> plain sum, scale at flush
            for (int t = 0; t < 4; ++t)
                for (int j = 0; j < 8; ++j)
                    s_acc[t][j] += uh[t][j];
        }
    }

    // ---- flush partial s via fp32 atomics (L2-resident 0.5 MB buffer) ----
    const float scale = (pass == 0) ? (1.0f / 16.0f) : 1.0f;
    for (int t = 0; t < 4; ++t)
        for (int j = 0; j < 8; ++j) {
            const int b = bt * 16 + half * 8 + j;        // C/D: VGPR j <-> M=j+8*half
            const int o = t * 16 + nl;
            atomicAdd(&s[(size_t)b * Ncol + wave * On + o], s_acc[t][j] * scale);
        }
}

// ---------------------------------------------------------------- launch
extern "C" void kernel_launch(void* const* d_in, const int* in_sizes, int n_in,
                              void* d_out, int out_size, void* d_ws, size_t ws_size,
                              hipStream_t stream) {
    const float* x = (const float*)d_in[0];   // (128, 64, 1152)
    const float* W = (const float*)d_in[1];   // (1, 1152, 16, 64, 64)
    float* out = (float*)d_out;               // (128, 16, 64, 1) flat = 131072

    float* xT  = (float*)d_ws;                                  // 9,437,184 f
    float* bij = xT + (size_t)Bsz * Rn * ICn;                   // 2,359,296 f
    float* s   = bij + (size_t)Bsz * Rn * Cn;                   // 131,072 f
    float* v   = s + (size_t)Bsz * Cn * On;                     // 131,072 f

    transpose_x<<<dim3(Rn / 32, ICn / 32, Bsz), dim3(32, 8), 0, stream>>>(x, xT);
    zero_s<<<512, 256, 0, stream>>>(s);

    const dim3 rgrid(8, RCHUNKS);
    routing_pass<<<rgrid, 512, 0, stream>>>(xT, W, bij, s, v, 0);
    squash_k<<<512, 256, 0, stream>>>(s, v);
    routing_pass<<<rgrid, 512, 0, stream>>>(xT, W, bij, s, v, 1);
    squash_k<<<512, 256, 0, stream>>>(s, v);
    routing_pass<<<rgrid, 512, 0, stream>>>(xT, W, bij, s, v, 2);
    squash_k<<<512, 256, 0, stream>>>(s, out);
}
